// W4A16NemotronHUniQLMamba2Mixer_83416854823032
// MI455X (gfx1250) — matrix operand order repaired and checked by reading, not compiled, not run on hardware
//
#include <hip/hip_runtime.h>
#include <hip/hip_bf16.h>
#include <math.h>
#include <stdint.h>

#define DEV __device__ __forceinline__

typedef __bf16 bf16;
typedef bf16  v16bf __attribute__((ext_vector_type(16)));
typedef float v8f   __attribute__((ext_vector_type(8)));

// ---- problem constants (match the reference) ----
constexpr int Hh    = 64;     // heads
constexpr int Pp    = 64;     // head dim
constexpr int Nst   = 128;    // state dim
constexpr int Gg    = 8;      // groups
constexpr int Dm    = 2048;   // d_model
constexpr int CSz   = 256;    // chunk size
constexpr int Ll    = 4096;   // sequence length
constexpr int INTER_ = 4096;  // H*P
constexpr int CONV_  = 6144;  // INTER + 2*G*N
constexpr int PROJ_  = 10304; // INTER + CONV + H
constexpr int NCh   = 16;     // L / CS
constexpr float EPSf = 1e-5f;

// ---- WMMA fragment index helpers (CDNA5 ISA 7.12.2, 16-bit A 16x32 / B 32x16) ----
DEV int a_col(int e, int lane) { return (e & 7) + ((e >> 3) << 4) + ((lane >> 4) << 3); }
DEV int b_row(int e, int lane) { return (e & 7) + ((e >> 3) << 4) + ((lane >> 4) << 3); }

DEV v8f wmma_bf16(v16bf a, v16bf b, v8f c) {
  return __builtin_amdgcn_wmma_f32_16x16x32_bf16(false, a, false, b, (short)0, c,
                                                 false, false);
}

// ---- CDNA5 async global->LDS copy (ASYNCcnt-tracked; ISA 10. / 15.18.3) ----
DEV void async_ld_b128(void* lds_ptr, const void* gptr) {
  unsigned lds_off = (unsigned)(uintptr_t)lds_ptr;
  unsigned long long ga = (unsigned long long)(uintptr_t)gptr;
  asm volatile("global_load_async_to_lds_b128 %0, %1, off"
               :: "v"(lds_off), "v"(ga) : "memory");
}
DEV void wait_async0() {
  asm volatile("s_wait_asynccnt 0" ::: "memory");
}

DEV float sigmoidf_(float x) { return 1.f / (1.f + expf(-x)); }
DEV float siluf_(float x)    { return x * sigmoidf_(x); }
DEV float softplusf_(float x){ return (x > 20.f) ? x : log1pf(expf(x)); }

// =====================================================================
// K1/K8: C[M,N] = A[M,K] * W[N,K]^T   (fp32 in, bf16 WMMA, fp32 out)
// Double-buffered async global->LDS staging; fully unrolled stage with
// per-thread base pointers + uniform strides.
// =====================================================================
constexpr int TS = 36; // padded LDS row stride in floats (144B: 16B-aligned)

__global__ __launch_bounds__(256)
void gemm_xWt(const float* __restrict__ A, const float* __restrict__ W,
              float* __restrict__ C, int M, int N, int Kd)
{
  __shared__ float Af[2][128 * TS];
  __shared__ float Wf[2][64 * TS];
  const int tid  = threadIdx.x;
  const int lane = tid & 31;
  const int wv   = tid >> 5;
  const int wm   = wv >> 1;   // 0..3
  const int wn   = wv & 1;    // 0..1
  const int m0 = blockIdx.y * 128;
  const int n0 = blockIdx.x * 64;
  const int sr = tid >> 3;    // 0..31 staging row
  const int sch = (tid & 7) * 4; // 16B chunk within row

  auto stage = [&](int b, int k0) {
    const float* ga = A + (size_t)(m0 + sr) * Kd + k0 + sch;
    float* la = &Af[b][sr * TS + sch];
#pragma unroll
    for (int i = 0; i < 4; ++i)            // rows sr + 32*i  (0..127)
      async_ld_b128(la + i * 32 * TS, ga + (size_t)i * 32 * Kd);
    const float* gw = W + (size_t)(n0 + sr) * Kd + k0 + sch;
    float* lw = &Wf[b][sr * TS + sch];
#pragma unroll
    for (int i = 0; i < 2; ++i)            // rows sr + 32*i  (0..63)
      async_ld_b128(lw + i * 32 * TS, gw + (size_t)i * 32 * Kd);
  };

  v8f acc[2][2] = {};
  stage(0, 0);

  int b = 0;
  for (int k0 = 0; k0 < Kd; k0 += 32, b ^= 1) {
    wait_async0();       // my wave's copies into buffer b are done
    __syncthreads();     // all waves' copies done; last iter's compute done
    if (k0 + 32 < Kd) stage(b ^ 1, k0 + 32);  // overlap next copy w/ compute

    v16bf af[2], bw[2];
#pragma unroll
    for (int mi = 0; mi < 2; ++mi) {
      int row = wm * 32 + mi * 16 + (lane & 15);
#pragma unroll
      for (int e = 0; e < 16; ++e)
        af[mi][e] = (bf16)Af[b][row * TS + a_col(e, lane)];
    }
#pragma unroll
    for (int ni = 0; ni < 2; ++ni) {
      int col = wn * 32 + ni * 16 + (lane & 15);
#pragma unroll
      for (int e = 0; e < 16; ++e)
        bw[ni][e] = (bf16)Wf[b][col * TS + b_row(e, lane)];
    }
#pragma unroll
    for (int mi = 0; mi < 2; ++mi)
#pragma unroll
      for (int ni = 0; ni < 2; ++ni)
        acc[mi][ni] = wmma_bf16(af[mi], bw[ni], acc[mi][ni]);
  }

#pragma unroll
  for (int mi = 0; mi < 2; ++mi)
#pragma unroll
    for (int ni = 0; ni < 2; ++ni) {
      int rbase = m0 + wm * 32 + mi * 16 + ((lane >> 4) << 3);
      int col   = n0 + wn * 32 + ni * 16 + (lane & 15);
#pragma unroll
      for (int r = 0; r < 8; ++r)
        C[(size_t)(rbase + r) * N + col] = acc[mi][ni][r];
    }
}

// =====================================================================
// K2: depthwise causal conv(K=4) + bias + SiLU over hbc slice of proj
// =====================================================================
__global__ __launch_bounds__(256)
void conv_silu(const float* __restrict__ proj, const float* __restrict__ cw,
               const float* __restrict__ cb, float* __restrict__ hbcv)
{
  size_t idx = (size_t)blockIdx.x * 256 + threadIdx.x; // L*CONV_
  int ch = (int)(idx % CONV_);
  int l  = (int)(idx / CONV_);
  float acc = cb[ch];
#pragma unroll
  for (int k = 0; k < 4; ++k) {
    int ls = l + k - 3;
    if (ls >= 0)
      acc += proj[(size_t)ls * PROJ_ + INTER_ + ch] * cw[ch * 4 + k];
  }
  hbcv[idx] = siluf_(acc);
}

// =====================================================================
// K2b: dt = softplus(proj_dt + dt_bias)
// =====================================================================
__global__ __launch_bounds__(256)
void dt_softplus(const float* __restrict__ proj, const float* __restrict__ dtb,
                 float* __restrict__ dtsp)
{
  int idx = blockIdx.x * 256 + threadIdx.x; // L*H
  int h = idx & 63;
  int l = idx >> 6;
  float v = proj[(size_t)l * PROJ_ + INTER_ + CONV_ + h] + dtb[h];
  dtsp[idx] = softplusf_(v);
}

// =====================================================================
// K3: per (head, chunk) inclusive cumsum of Abar = dt * (-exp(A_log))
// =====================================================================
__global__ __launch_bounds__(256)
void chunk_cumsum(const float* __restrict__ dtsp, const float* __restrict__ A_log,
                  float* __restrict__ acs, float* __restrict__ cdec)
{
  int h = blockIdx.x >> 4;
  int c = blockIdx.x & 15;
  int tid = threadIdx.x;
  int l = c * CSz + tid;
  float Ah = -expf(A_log[h]);
  float v  = dtsp[(size_t)l * Hh + h] * Ah;
  __shared__ float buf[256];
  buf[tid] = v;
  __syncthreads();
  for (int off = 1; off < 256; off <<= 1) {
    float t = (tid >= off) ? buf[tid - off] : 0.f;
    __syncthreads();
    buf[tid] += t;
    __syncthreads();
  }
  acs[(size_t)h * Ll + l] = buf[tid];
  if (tid == 255) cdec[h * NCh + c] = expf(buf[255]);
}

// =====================================================================
// K4: fused intra-chunk SSM per (head, chunk):
//   scores = C*B^T (WMMA), mask/decay, Y_diag = scores*xdt (WMMA),
//   states = (xdt*decay)^T * B (WMMA).
// B_c async-staged to dynamic LDS in 2 phases of 128 rows.
// =====================================================================
constexpr int BST = 132; // B-stage row stride (floats), 528B: 16B-aligned

__global__ __launch_bounds__(256)
void chunk_ssm(const float* __restrict__ hbcv, const float* __restrict__ dtsp,
               const float* __restrict__ acs, const float* __restrict__ Dp,
               float* __restrict__ Ybuf, float* __restrict__ states)
{
  const int blk = blockIdx.x;
  const int h = blk >> 4;     // / NCh
  const int c = blk & 15;
  const int g = h >> 3;       // H/G = 8
  const int tid = threadIdx.x, lane = tid & 31, wv = tid >> 5;

  __shared__ bf16  s_xdt[256][66];      // [s][p], padded
  __shared__ float s_acs[256];
  __shared__ float s_dec[256];
  __shared__ float s_stage[8][16 * 32]; // per-wave masked-score staging
  extern __shared__ __attribute__((aligned(16))) float s_B[]; // 128 x BST f32

  const size_t rowBase = (size_t)c * CSz;
  const float* __restrict__ Cbase = hbcv + rowBase * CONV_ + INTER_ + Gg * Nst + g * Nst;
  const float* __restrict__ Bbase = hbcv + rowBase * CONV_ + INTER_ + g * Nst;

  // stage xdt = x * dt (bf16) and cumsum values
  for (int i = tid; i < 256 * 64; i += 256) {
    int s = i >> 6, p = i & 63;
    float xv = hbcv[(rowBase + s) * CONV_ + h * Pp + p];
    float dv = dtsp[(rowBase + s) * Hh + h];
    s_xdt[s][p] = (bf16)(xv * dv);
  }
  s_acs[tid] = acs[(size_t)h * Ll + rowBase + tid];
  __syncthreads();
  s_dec[tid] = expf(s_acs[255] - s_acs[tid]);   // decay_states
  __syncthreads();

  // hoist C-matrix A-fragments (l x n, K=128 -> 4 k-steps) for this wave
  v16bf aC[2][4];
#pragma unroll
  for (int li = 0; li < 2; ++li) {
    int lrow = wv * 32 + li * 16 + (lane & 15);
#pragma unroll
    for (int kk = 0; kk < 4; ++kk)
#pragma unroll
      for (int e = 0; e < 16; ++e)
        aC[li][kk][e] = (bf16)Cbase[(size_t)lrow * CONV_ + kk * 32 + a_col(e, lane)];
  }

  v8f Yacc[2][4] = {};
  v8f stacc[4]   = {};
  const int sr  = tid >> 5;        // 0..7 staging base row
  const int sch = (tid & 31) * 4;  // 16B chunk within 128-float row

  for (int ph = 0; ph < 2; ++ph) {
    __syncthreads();   // previous phase fully consumed before overwrite
    {
      const float* gb = Bbase + (size_t)(ph * 128 + sr) * CONV_ + sch;
      float* lb = &s_B[sr * BST + sch];
#pragma unroll
      for (int i = 0; i < 16; ++i)        // rows sr + 8*i (0..127)
        async_ld_b128(lb + i * 8 * BST, gb + (size_t)i * 8 * CONV_);
    }
    wait_async0();
    __syncthreads();

#pragma unroll
    for (int s32l = 0; s32l < 128; s32l += 32) {
      const int s32 = ph * 128 + s32l;
      // scores for both l-tiles and both s16 halves; bB loaded once per (sh,kk)
      v8f sc[2][2] = {};
#pragma unroll
      for (int sh = 0; sh < 2; ++sh) {
        int srow = s32l + sh * 16 + (lane & 15);
#pragma unroll
        for (int kk = 0; kk < 4; ++kk) {
          v16bf bB;
#pragma unroll
          for (int e = 0; e < 16; ++e)
            bB[e] = (bf16)s_B[srow * BST + kk * 32 + b_row(e, lane)];
#pragma unroll
          for (int li = 0; li < 2; ++li)
            sc[li][sh] = wmma_bf16(aC[li][kk], bB, sc[li][sh]);
        }
      }
#pragma unroll
      for (int li = 0; li < 2; ++li) {
        int lt = wv * 32 + li * 16;
#pragma unroll
        for (int sh = 0; sh < 2; ++sh) {
          int scol = s32 + sh * 16 + (lane & 15);
          float a_s = s_acs[scol];
          int r0 = lt + ((lane >> 4) << 3);
#pragma unroll
          for (int r = 0; r < 8; ++r) {
            int lrow = r0 + r;
            float v = (scol <= lrow) ? sc[li][sh][r] * expf(s_acs[lrow] - a_s) : 0.f;
            s_stage[wv][(((lane >> 4) << 3) + r) * 32 + sh * 16 + (lane & 15)] = v;
          }
        }
        // re-load staged scores as bf16 A-fragment (16 x 32 over s)
        v16bf aS;
#pragma unroll
        for (int e = 0; e < 16; ++e)
          aS[e] = (bf16)s_stage[wv][(lane & 15) * 32 + a_col(e, lane)];
#pragma unroll
        for (int pi = 0; pi < 4; ++pi) {
          v16bf bX;
          int pcol = pi * 16 + (lane & 15);
#pragma unroll
          for (int e = 0; e < 16; ++e)
            bX[e] = s_xdt[s32 + b_row(e, lane)][pcol];
          Yacc[li][pi] = wmma_bf16(aS, bX, Yacc[li][pi]);
        }
      }
      // states partial sums for this s32 block (reads staged B)
#pragma unroll
      for (int j = 0; j < 4; ++j) {
        int t  = wv * 4 + j;     // 0..31 over 4x8 tile grid
        int pi = t >> 3, ni = t & 7;
        int prow = pi * 16 + (lane & 15);
        int ncol = ni * 16 + (lane & 15);
        v16bf aX, bBm;
#pragma unroll
        for (int e = 0; e < 16; ++e) {
          int s = s32 + a_col(e, lane);
          aX[e] = (bf16)((float)s_xdt[s][prow] * s_dec[s]);
        }
#pragma unroll
        for (int e = 0; e < 16; ++e) {
          int sl = s32l + b_row(e, lane);
          bBm[e] = (bf16)s_B[sl * BST + ncol];
        }
        stacc[j] = wmma_bf16(aX, bBm, stacc[j]);
      }
    }
  }

  // write Y_diag + D*x
  const float Dh = Dp[h];
#pragma unroll
  for (int li = 0; li < 2; ++li)
#pragma unroll
    for (int pi = 0; pi < 4; ++pi) {
      int r0   = wv * 32 + li * 16 + ((lane >> 4) << 3);
      int pcol = pi * 16 + (lane & 15);
#pragma unroll
      for (int r = 0; r < 8; ++r) {
        int lrow = r0 + r;
        float xv = hbcv[(rowBase + lrow) * CONV_ + h * Pp + pcol];
        Ybuf[(rowBase + lrow) * INTER_ + h * Pp + pcol] = Yacc[li][pi][r] + Dh * xv;
      }
    }

  // write states[c,h,p,n]
#pragma unroll
  for (int j = 0; j < 4; ++j) {
    int t  = wv * 4 + j;
    int pi = t >> 3, ni = t & 7;
    int ncol = ni * 16 + (lane & 15);
    int pr0  = pi * 16 + ((lane >> 4) << 3);
#pragma unroll
    for (int r = 0; r < 8; ++r)
      states[(((size_t)c * Hh + h) * Pp + pr0 + r) * Nst + ncol] = stacc[j][r];
  }
}

// =====================================================================
// K5: sequential 16-step inter-chunk recurrence (per h,p,n element)
// =====================================================================
__global__ __launch_bounds__(256)
void chunk_scan_k(const float* __restrict__ states, const float* __restrict__ cdec,
                  float* __restrict__ prevs)
{
  int idx = blockIdx.x * 256 + threadIdx.x; // H*P*N = 524288
  int h   = idx >> 13;                      // /(P*N)
  int rem = idx & 8191;
  float acc = 0.f;
#pragma unroll
  for (int c = 0; c < NCh; ++c) {
    size_t o = ((size_t)c * Hh + h) * 8192 + rem;
    prevs[o] = acc;
    acc = acc * cdec[h * NCh + c] + states[o];
  }
}

// =====================================================================
// K6: Y += state_decay[l] * C * prev_states^T   (256x64, K=128, WMMA)
// prev_states tile (64x128 f32, 32KB) async-copied to LDS once per block.
// =====================================================================
constexpr int PSL = 132; // padded LDS row stride (floats), 16B-aligned rows

__global__ __launch_bounds__(256)
void yoff_add(const float* __restrict__ hbcv, const float* __restrict__ prevs,
              const float* __restrict__ acs, float* __restrict__ Ybuf)
{
  const int blk = blockIdx.x;
  const int h = blk >> 4;
  const int c = blk & 15;
  const int g = h >> 3;
  const int tid = threadIdx.x, lane = tid & 31, wv = tid >> 5;
  const size_t rowBase = (size_t)c * CSz;
  const float* __restrict__ Cbase = hbcv + rowBase * CONV_ + INTER_ + Gg * Nst + g * Nst;
  const float* __restrict__ Pv = prevs + ((size_t)c * Hh + h) * (size_t)(Pp * Nst);

  __shared__ float s_prev[Pp * PSL]; // 64 rows x 132 floats (~33.8KB)
  {
    const int p0 = tid >> 5;          // 0..7
    const int ch = (tid & 31) * 4;    // 16B chunk in 128-float row
    const float* gp = Pv + (size_t)p0 * Nst + ch;
    float* lp = &s_prev[p0 * PSL + ch];
#pragma unroll
    for (int i = 0; i < 8; ++i)       // rows p0 + 8*i (0..63)
      async_ld_b128(lp + i * 8 * PSL, gp + (size_t)i * 8 * Nst);
  }
  wait_async0();
  __syncthreads();

#pragma unroll
  for (int li = 0; li < 2; ++li) {
    int lrow16 = wv * 32 + li * 16 + (lane & 15);
    v16bf aC[4];
#pragma unroll
    for (int kk = 0; kk < 4; ++kk)
#pragma unroll
      for (int e = 0; e < 16; ++e)
        aC[kk][e] = (bf16)Cbase[(size_t)lrow16 * CONV_ + kk * 32 + a_col(e, lane)];
#pragma unroll
    for (int pi = 0; pi < 4; ++pi) {
      v8f acc = {};
      int pcol = pi * 16 + (lane & 15);
#pragma unroll
      for (int kk = 0; kk < 4; ++kk) {
        v16bf bP;
#pragma unroll
        for (int e = 0; e < 16; ++e)
          bP[e] = (bf16)s_prev[pcol * PSL + kk * 32 + b_row(e, lane)];
        acc = wmma_bf16(aC[kk], bP, acc);
      }
      int r0 = wv * 32 + li * 16 + ((lane >> 4) << 3);
#pragma unroll
      for (int r = 0; r < 8; ++r) {
        int lrow = r0 + r;
        float sd = expf(acs[(size_t)h * Ll + rowBase + lrow]);
        size_t yi = (rowBase + lrow) * INTER_ + h * Pp + pcol;
        Ybuf[yi] += acc[r] * sd;
      }
    }
  }
}

// =====================================================================
// K7: gate (y * silu(gate)) + grouped RMS norm (group=512) * norm_weight
// =====================================================================
__global__ __launch_bounds__(256)
void gate_norm(const float* __restrict__ Ybuf, const float* __restrict__ proj,
               const float* __restrict__ nw, float* __restrict__ normY)
{
  __shared__ float red[256];
  int l   = blockIdx.x >> 3;
  int grp = blockIdx.x & 7;
  int tid = threadIdx.x;
  size_t ybase = (size_t)l * INTER_ + grp * 512;
  size_t gbase = (size_t)l * PROJ_ + grp * 512;
  float gv[2];
  float ssq = 0.f;
#pragma unroll
  for (int j = 0; j < 2; ++j) {
    int idx = tid + j * 256;
    float y  = Ybuf[ybase + idx];
    float gt = proj[gbase + idx];
    float v  = y * siluf_(gt);
    gv[j] = v;
    ssq += v * v;
  }
  red[tid] = ssq;
  __syncthreads();
  for (int o = 128; o > 0; o >>= 1) {
    if (tid < o) red[tid] += red[tid + o];
    __syncthreads();
  }
  float scale = rsqrtf(red[0] / 512.f + EPSf);
#pragma unroll
  for (int j = 0; j < 2; ++j) {
    int idx = tid + j * 256;
    normY[ybase + idx] = gv[j] * scale * nw[grp * 512 + idx];
  }
}

// =====================================================================
extern "C" void kernel_launch(void* const* d_in, const int* in_sizes, int n_in,
                              void* d_out, int out_size, void* d_ws, size_t ws_size,
                              hipStream_t stream)
{
  (void)in_sizes; (void)n_in; (void)out_size; (void)ws_size;
  const float* hid     = (const float*)d_in[0];
  const float* W_in    = (const float*)d_in[1];
  const float* conv_w  = (const float*)d_in[2];
  const float* conv_b  = (const float*)d_in[3];
  const float* dt_bias = (const float*)d_in[4];
  const float* A_log   = (const float*)d_in[5];
  const float* Dvec    = (const float*)d_in[6];
  const float* norm_w  = (const float*)d_in[7];
  const float* W_out   = (const float*)d_in[8];
  float* out = (float*)d_out;

  float* ws = (float*)d_ws;
  size_t off = 0;
  float* proj   = ws + off; off += (size_t)Ll * PROJ_;
  float* hbcv   = ws + off; off += (size_t)Ll * CONV_;
  float* dtsp   = ws + off; off += (size_t)Ll * Hh;
  float* acs    = ws + off; off += (size_t)Hh * Ll;
  float* cdec   = ws + off; off += (size_t)Hh * NCh;
  float* states = ws + off; off += (size_t)NCh * Hh * Pp * Nst;
  float* prevs  = ws + off; off += (size_t)NCh * Hh * Pp * Nst;
  float* Ybuf   = ws + off; off += (size_t)Ll * INTER_;
  float* normY  = ws + off; off += (size_t)Ll * INTER_;

  dim3 blk(256);
  const size_t bDyn = (size_t)128 * BST * sizeof(float); // 67584B dynamic LDS
  gemm_xWt<<<dim3(PROJ_ / 64, Ll / 128), blk, 0, stream>>>(hid, W_in, proj, Ll, PROJ_, Dm);
  conv_silu<<<dim3((Ll * CONV_) / 256), blk, 0, stream>>>(proj, conv_w, conv_b, hbcv);
  dt_softplus<<<dim3((Ll * Hh) / 256), blk, 0, stream>>>(proj, dt_bias, dtsp);
  chunk_cumsum<<<dim3(Hh * NCh), blk, 0, stream>>>(dtsp, A_log, acs, cdec);
  chunk_ssm<<<dim3(Hh * NCh), blk, bDyn, stream>>>(hbcv, dtsp, acs, Dvec, Ybuf, states);
  chunk_scan_k<<<dim3((Hh * Pp * Nst) / 256), blk, 0, stream>>>(states, cdec, prevs);
  yoff_add<<<dim3(Hh * NCh), blk, 0, stream>>>(hbcv, prevs, acs, Ybuf);
  gate_norm<<<dim3(Ll * Gg), blk, 0, stream>>>(Ybuf, proj, norm_w, normY);
  gemm_xWt<<<dim3(Dm / 64, Ll / 128), blk, 0, stream>>>(normY, W_out, out, Ll, Dm, INTER_);
}
